// EEG_Double_28355374088684
// MI455X (gfx1250) — matrix operand order repaired and checked by reading, not compiled, not run on hardware
//
#include <hip/hip_runtime.h>
#include <stdint.h>

typedef _Float16 v8h  __attribute__((ext_vector_type(8)));
typedef _Float16 v16h __attribute__((ext_vector_type(16)));
typedef float    v8f  __attribute__((ext_vector_type(8)));
typedef uint32_t v4u  __attribute__((ext_vector_type(4)));
typedef int      v4i  __attribute__((ext_vector_type(4)));
typedef int      v8i  __attribute__((ext_vector_type(8)));

#define B_TOTAL 4096
#define C_IN    14
#define T_STEPS 600
#define O_OUT   500
#define BT      16          // batch rows per workgroup
#define TB      60          // timesteps staged per LDS block (600 % 60 == 0)
#define TILES_PER_WAVE 4    // 8 waves * 4 tiles * 16 = 512 output channels >= 500
#define VTH     7.0f

static __device__ inline v8f v8f_zero() {
    v8f z = {0.f, 0.f, 0.f, 0.f, 0.f, 0.f, 0.f, 0.f};
    return z;
}
static __device__ inline v16h v16h_zero() {
    v16h z = {(_Float16)0, (_Float16)0, (_Float16)0, (_Float16)0,
              (_Float16)0, (_Float16)0, (_Float16)0, (_Float16)0,
              (_Float16)0, (_Float16)0, (_Float16)0, (_Float16)0,
              (_Float16)0, (_Float16)0, (_Float16)0, (_Float16)0};
    return z;
}

__global__ __launch_bounds__(256)
void snn_lif_wmma_kernel(const float* __restrict__ x,
                         const float* __restrict__ W1,
                         const float* __restrict__ W2,
                         float* __restrict__ out)
{
    // x tile staged by the Tensor Data Mover: [m=16][k=14][tt=TB] f32
    __shared__ __align__(16) float    xs[BT * C_IN * TB];
    // spike matrix in WMMA A staging form: 16 rows x 32 K (K=14..31 stay zero)
    __shared__ __align__(16) _Float16 s1a[BT * 32];

    const int tid  = threadIdx.x;
    const int lane = tid & 31;
    const int wave = tid >> 5;
    const int b0   = blockIdx.x * BT;

    // zero the spike staging buffer once (padding region is never rewritten)
    for (int i = tid; i < BT * 32; i += 256) s1a[i] = (_Float16)0.0f;

    // ---- block-1 per-thread state: thread (m, j), m = batch row, j = neuron ----
    const int  m   = tid / C_IN;
    const int  j   = tid % C_IN;
    const bool doA = (tid < BT * C_IN);   // waves 0..6 fully active, wave 7 idle
    float w1row[C_IN];
    float v1 = 0.0f, cache = 0.0f;
    if (doA) {
        #pragma unroll
        for (int k = 0; k < C_IN; ++k) w1row[k] = W1[j * C_IN + k];
    }

    // ---- preload W2 B-fragments once (constant over all 600 steps) ----
    // B layout used here (contiguous K, same as A): lanes 0-15 hold N=lane, K=0..15 (2 per VGPR);
    // lanes 16-31 hold K=16..31 -> all zero here since K(valid) < 14.
    const int n    = lane & 15;            // output column within tile
    const int madd = (lane >> 4) << 3;     // C/D row offset: lanes>=16 -> M+8
    v16h bfrag[TILES_PER_WAVE];
    int  obase[TILES_PER_WAVE];
    #pragma unroll
    for (int i = 0; i < TILES_PER_WAVE; ++i) {
        obase[i] = (wave * TILES_PER_WAVE + i) * 16;
        v16h b = v16h_zero();
        const int o = obase[i] + n;
        if (lane < 16 && o < O_OUT) {
            #pragma unroll
            for (int k = 0; k < C_IN; ++k) b[k] = (_Float16)W2[o * C_IN + k];
        }
        bfrag[i] = b;
    }

    // membrane state v2, kept in WMMA C/D register layout for the whole scan
    v8f vacc[TILES_PER_WAVE];
    #pragma unroll
    for (int i = 0; i < TILES_PER_WAVE; ++i) vacc[i] = v8f_zero();

    for (int t0 = 0; t0 < T_STEPS; t0 += TB) {
        // ---- stage x[b0:b0+16, :, t0:t0+TB) into LDS ----
#if defined(__gfx1250__) && __has_builtin(__builtin_amdgcn_tensor_load_to_lds)
        if (wave == 0) {
            // 3D TDM tile: dim0 = TB (t, contiguous), dim1 = 14 (c, stride 600),
            // dim2 = 16 (b, stride 14*600). data_size = 4 bytes.
            const uint64_t ga = (uint64_t)(uintptr_t)x +
                                4ull * ((uint64_t)b0 * (C_IN * T_STEPS) + (uint64_t)t0);
            const uint32_t lds_off = (uint32_t)(uintptr_t)(void*)xs; // low 32b of LDS aperture addr
            v4u g0;
            g0.x = 1u;                                              // count=1, user mode
            g0.y = lds_off;                                         // lds_addr
            g0.z = (uint32_t)(ga & 0xFFFFFFFFu);                    // global_addr[31:0]
            g0.w = (uint32_t)((ga >> 32) & 0x1FFFFFFu) | (2u << 30); // addr[56:32] | type=2
            v8i g1;
            g1[0] = (2 << 16);                          // wg_mask=0, data_size=4B
            g1[1] = (int)((TB & 0xFFFF) << 16);         // tensor_dim0 lo16 @ bits[63:48]
            g1[2] = (int)((C_IN & 0xFFFF) << 16);       // dim0 hi16=0 | tensor_dim1 lo16
            g1[3] = (int)((TB & 0xFFFF) << 16);         // dim1 hi16=0 | tile_dim0=TB
            g1[4] = (int)(C_IN | (BT << 16));           // tile_dim1=14 | tile_dim2=16
            g1[5] = (int)T_STEPS;                       // tensor_dim0_stride lo32 = 600
            g1[6] = (int)(((C_IN * T_STEPS) & 0xFFFF) << 16); // stride0 hi=0 | stride1 lo16
            g1[7] = (int)((C_IN * T_STEPS) >> 16);      // tensor_dim1_stride bits[47:16]
            v4i g2; g2[0] = BT; g2[1] = 0; g2[2] = 0; g2[3] = 0;   // tensor_dim2=16, dim3 unused
            v4i g3; g3[0] = 0;  g3[1] = 0; g3[2] = 0; g3[3] = 0;
            v8i g4; g4[0] = 0;  g4[1] = 0; g4[2] = 0; g4[3] = 0;
                    g4[4] = 0;  g4[5] = 0; g4[6] = 0; g4[7] = 0;   // extra group (6-arg form)
            __builtin_amdgcn_tensor_load_to_lds(g0, g1, g2, g3, g4, 0);
            __builtin_amdgcn_s_wait_tensorcnt(0);
        }
#else
        for (int idx = tid; idx < BT * C_IN * TB; idx += 256) {
            const int mm = idx / (C_IN * TB);
            const int r  = idx % (C_IN * TB);
            const int kk = r / TB;
            const int tt = r % TB;
            xs[idx] = x[((size_t)(b0 + mm) * C_IN + kk) * T_STEPS + t0 + tt];
        }
#endif
        __syncthreads();

        for (int tt = 0; tt < TB; ++tt) {
            // ---- block 1: 14x14 recurrent linear + LIF, one thread per (m, j) ----
            if (doA) {
                float acc = 0.5f * cache;                 // decayed spike feedback
                #pragma unroll
                for (int k = 0; k < C_IN; ++k)
                    acc += xs[(m * C_IN + k) * TB + tt] * w1row[k];
                v1 = 0.5f * (v1 + acc);                   // v += (i - v)/tau, tau=2
                const bool fire = (v1 >= VTH);
                cache = fire ? 1.0f : 0.0f;               // s1 (next-step feedback)
                v1    = fire ? 0.0f : v1;                 // hard reset
                s1a[m * 32 + j] = (_Float16)cache;
            }
            __syncthreads();

            // ---- block 2: s1 @ W2.T via WMMA f16 (K padded 14 -> 32) ----
            // A fragment (16-bit 16x32): lane slice K = kbase..kbase+15 (contiguous)
            const int arow  = lane & 15;
            const int kbase = (lane & 16) ? 16 : 0;
            const v8h* ap = (const v8h*)&s1a[arow * 32 + kbase];
            const v8h lo = ap[0];
            const v8h hi = ap[1];        // +8 halves
            v16h a;
            #pragma unroll
            for (int i = 0; i < 8; ++i) { a[i] = lo[i]; a[i + 8] = hi[i]; }

            #pragma unroll
            for (int i = 0; i < TILES_PER_WAVE; ++i) {
                v8f c = v8f_zero();
                c = __builtin_amdgcn_wmma_f32_16x16x32_f16(
                        false, a, false, bfrag[i], (short)0, c, false, false);
                // LIF update on the 16x16 i2 tile, state stays in registers
                #pragma unroll
                for (int e = 0; e < 8; ++e) {
                    const float v = 0.5f * (vacc[i][e] + c[e]);
                    vacc[i][e] = (v >= VTH) ? 0.0f : v;
                }
            }
            __syncthreads();   // s1a may be rewritten next step
        }
    }

    // ---- epilogue: out[b, o] = exp(v2) ----
    #pragma unroll
    for (int i = 0; i < TILES_PER_WAVE; ++i) {
        const int o = obase[i] + n;
        if (o < O_OUT) {
            #pragma unroll
            for (int r = 0; r < 8; ++r) {
                const int bb = b0 + madd + r;       // C/D row M = r (+8 for lanes>=16)
                out[(size_t)bb * O_OUT + o] = __expf(vacc[i][r]);
            }
        }
    }
}

extern "C" void kernel_launch(void* const* d_in, const int* in_sizes, int n_in,
                              void* d_out, int out_size, void* d_ws, size_t ws_size,
                              hipStream_t stream) {
    const float* x  = (const float*)d_in[0];   // [4096, 14, 600]
    const float* W1 = (const float*)d_in[1];   // [14, 14]
    const float* W2 = (const float*)d_in[2];   // [500, 14]
    float* out = (float*)d_out;                // [4096, 500]
    (void)in_sizes; (void)n_in; (void)out_size; (void)d_ws; (void)ws_size;

    dim3 grid(B_TOTAL / BT);   // 256 workgroups, one 16-batch tile each
    dim3 block(256);           // 8 wave32 waves
    snn_lif_wmma_kernel<<<grid, block, 0, stream>>>(x, W1, W2, out);
}
